// QuantLinear_37890201485500
// MI455X (gfx1250) — matrix operand order repaired and checked by reading, not compiled, not run on hardware
//
#include <hip/hip_runtime.h>

typedef __attribute__((ext_vector_type(16))) _Float16 v16h;
typedef __attribute__((ext_vector_type(8)))  float    v8f;

constexpr int K_DIM = 4096;
constexpr int N_DIM = 11008;

constexpr int BM = 128;
constexpr int BN = 256;
constexpr int BK = 64;
constexpr int WAVES_N = 4;   // 2x4 wave grid over the 128x256 block
constexpr int TM = 4;        // 4 M-tiles of 16 per wave  (64 rows)
constexpr int TN = 4;        // 4 N-tiles of 16 per wave  (64 cols)

__global__ __launch_bounds__(256)
void QuantLinear_w4a16_wmma(const float* __restrict__ x,
                            const int*   __restrict__ wq,
                            const float* __restrict__ scale,
                            const float* __restrict__ bias,
                            float* __restrict__ y)
{
    // Fragment-native LDS layout: [tile16][kstep32][lane][slot]
    // Each lane's fragment is 32 contiguous bytes -> 2x ds_load_b128
    __shared__ __align__(32) _Float16 lA[BM / 16][BK / 32][32][16];  // 16 KB
    __shared__ __align__(32) _Float16 lB[BN / 16][BK / 32][32][16];  // 32 KB

    const int tid  = threadIdx.x;
    const int lane = tid & 31;
    const int wave = tid >> 5;
    const int wm   = wave / WAVES_N;   // 0..1
    const int wn   = wave % WAVES_N;   // 0..3

    const int blockN = blockIdx.x * BN;
    const int blockM = blockIdx.y * BM;

    v8f acc[TM][TN] = {};

    for (int k0 = 0; k0 < K_DIM; k0 += BK) {
        if (k0 + BK < K_DIM) {
            // prefetch next K-tiles (emits global_prefetch_b8)
            __builtin_prefetch(x  + (size_t)(blockM + (tid >> 1)) * K_DIM     + (k0 + BK)        + ((tid & 1) << 5), 0, 1);
            __builtin_prefetch(wq + (size_t)(blockN +  tid      ) * (K_DIM/2) + ((k0 + BK) >> 1), 0, 1);
        }

        // ---- stage A: 128x64 fp32 -> f16 fragments (2048 float4 loads) ----
        #pragma unroll
        for (int it = 0; it < 8; ++it) {
            const int i      = tid + it * 256;
            const int mLocal = i >> 4;            // 0..127
            const int kkBase = (i & 15) << 2;     // 0..60 step 4
            const float4 v = *(const float4*)(x + (size_t)(blockM + mLocal) * K_DIM + k0 + kkBase);
            const int tileM = mLocal >> 4;
            const int r     = mLocal & 15;
            const float f[4] = {v.x, v.y, v.z, v.w};
            #pragma unroll
            for (int j = 0; j < 4; ++j) {
                const int kk   = kkBase + j;
                const int ks   = kk >> 5;                       // which 16x32 k-step
                const int w32  = kk & 31;
                const int g    = w32 & 15;
                const int ln   = (g < 8) ? r : (r + 16);        // ISA A-layout lane
                const int slot = ((w32 >> 4) << 3) + (w32 & 7); // ISA A-layout slot
                lA[tileM][ks][ln][slot] = (_Float16)f[j];
            }
        }

        // ---- stage B: 256 rows x 32 packed nibble-bytes (as int32) ----
        #pragma unroll
        for (int it = 0; it < 8; ++it) {
            const int i      = tid + it * 256;
            const int nLocal = i >> 3;            // 0..255
            const int j4     = (i & 7) << 2;      // byte index base 0..28
            const int4 b4 = *(const int4*)(wq + (size_t)(blockN + nLocal) * (K_DIM / 2) + (k0 >> 1) + j4);
            const int tileN = nLocal >> 4;
            const int nr    = nLocal & 15;
            const int bs[4] = {b4.x, b4.y, b4.z, b4.w};
            #pragma unroll
            for (int j = 0; j < 4; ++j) {
                const unsigned byte = (unsigned)bs[j] & 0xFFu;
                const int lo = (int)(byte & 0xFu) - 8;          // even k
                const int hi = (int)((byte >> 4) & 0xFu) - 8;   // odd k
                const int kk   = (j4 + j) << 1;                 // 0..62 even
                const int ks   = kk >> 5;
                const int w32  = kk & 31;
                const int ln   = ((w32 & 15) < 8) ? nr : (nr + 16);   // ISA B-layout lane
                const int slot = ((w32 >> 4) << 3) + (w32 & 7);
                const _Float16 hlo = (_Float16)(float)lo;
                const _Float16 hhi = (_Float16)(float)hi;
                const unsigned packed = ((unsigned)__builtin_bit_cast(unsigned short, hhi) << 16)
                                      |  (unsigned)__builtin_bit_cast(unsigned short, hlo);
                *(unsigned*)&lB[tileN][ks][ln][slot] = packed;  // slot is even -> dword aligned
            }
        }
        __syncthreads();

        // ---- compute: 2 k-steps x (4x4) tiles = 32 WMMAs per iteration ----
        #pragma unroll
        for (int ks = 0; ks < BK / 32; ++ks) {
            v16h aF[TM];
            #pragma unroll
            for (int i = 0; i < TM; ++i)
                aF[i] = *(const v16h*)&lA[wm * TM + i][ks][lane][0];
            #pragma unroll
            for (int j = 0; j < TN; ++j) {
                const v16h bF = *(const v16h*)&lB[wn * TN + j][ks][lane][0];
                #pragma unroll
                for (int i = 0; i < TM; ++i)
                    acc[i][j] = __builtin_amdgcn_wmma_f32_16x16x32_f16(
                        false, aF[i], false, bF, (short)0, acc[i][j], false, false);
            }
        }
        __syncthreads();
    }

    // ---- epilogue: y = acc * scale[n] + bias[n] ----
    // C/D layout: lanes 0-15 -> N=lane, M=vgpr i; lanes 16-31 -> N=lane-16, M=i+8
    const int nCol   = lane & 15;
    const int rowOff = (lane >> 4) << 3;   // 0 or 8
    #pragma unroll
    for (int j = 0; j < TN; ++j) {
        const int o  = blockN + wn * 64 + j * 16 + nCol;
        const float s  = scale[o];
        const float bv = bias[o];
        #pragma unroll
        for (int i = 0; i < TM; ++i) {
            const int mBase = blockM + wm * 64 + i * 16 + rowOff;
            #pragma unroll
            for (int r = 0; r < 8; ++r) {
                y[(size_t)(mBase + r) * N_DIM + o] = acc[i][j][r] * s + bv;
            }
        }
    }
}

extern "C" void kernel_launch(void* const* d_in, const int* in_sizes, int n_in,
                              void* d_out, int out_size, void* d_ws, size_t ws_size,
                              hipStream_t stream) {
    const float* x     = (const float*)d_in[0];
    const int*   wq    = (const int*)d_in[1];
    const float* scale = (const float*)d_in[2];
    const float* bias  = (const float*)d_in[3];
    float*       y     = (float*)d_out;

    const int M = in_sizes[0] / K_DIM;     // 4*2048 = 8192
    dim3 grid(N_DIM / BN, M / BM);         // (43, 64)
    dim3 block(256);
    QuantLinear_w4a16_wmma<<<grid, block, 0, stream>>>(x, wq, scale, bias, y);
    (void)d_ws; (void)ws_size; (void)n_in; (void)out_size;
}
